// MultiDisplacerNet_5987184411089
// MI455X (gfx1250) — matrix operand order repaired and checked by reading, not compiled, not run on hardware
//
#include <hip/hip_runtime.h>
#include <hip/hip_bf16.h>

// ---------------------------------------------------------------------------
// MultiDisplacerNet on MI455X (gfx1250): WMMA-based dynamic-kNN GATv2 stack.
// Dominant cost = 4 full Gram matrices [3,8192,8192] -> v_wmma_f32_16x16x32_f16.
// GEMM k-loop: 2-stage A+B pipeline (both global). kNN k-loop: B-only pipeline
// (A is LDS-resident; saves 16 VGPRs -> avoids top-16 state spills).
// ---------------------------------------------------------------------------

typedef _Float16 f16;
typedef __attribute__((ext_vector_type(16))) _Float16 v16h;
typedef __attribute__((ext_vector_type(8)))  _Float16 v8h;
typedef __attribute__((ext_vector_type(8)))  float    v8f;

#define NB 3
#define NN 8192
#define KNN_K 16

// ---- WMMA fragment loaders (CDNA5 ISA 7.12.2 layouts, wave32) -------------
// A-fragment 16x32 f16: lane<16 holds row M=lane, K={k0..k0+7, k0+16..k0+23};
// lane>=16 holds row M=lane-16, K={k0+8..k0+15, k0+24..k0+31}.
__device__ __forceinline__ v16h load_fragA(const f16* base, int ld, int row0, int k0) {
  int lane = threadIdx.x & 31;
  int m = lane & 15, h = lane >> 4;
  const f16* p = base + (size_t)(row0 + m) * ld + k0 + h * 8;
  v8h lo = *(const v8h*)p;          // K = k0 + h*8 .. +7
  v8h hi = *(const v8h*)(p + 16);   // K = k0 + 16 + h*8 .. +7
  v16h r;
#pragma unroll
  for (int i = 0; i < 8; ++i) { r[i] = lo[i]; r[8 + i] = hi[i]; }
  return r;
}

// B-fragment 32x16 f16 loaded from the TRANSPOSED operand (row-major BT[n][k]):
// lane<16 holds col N=lane, K={k0..k0+7, k0+16..k0+23}; lane>=16 holds col N=lane-16, K={k0+8..k0+15, k0+24..k0+31}.
__device__ __forceinline__ v16h load_fragB(const f16* baseT, int ld, int n0, int k0) {
  int lane = threadIdx.x & 31;
  int n = lane & 15, h = lane >> 4;
  const f16* p = baseT + (size_t)(n0 + n) * ld + k0 + h * 8;
  v8h lo = *(const v8h*)p;
  v8h hi = *(const v8h*)(p + 16);
  v16h r;
#pragma unroll
  for (int i = 0; i < 8; ++i) { r[i] = lo[i]; r[8 + i] = hi[i]; }
  return r;
}

__device__ __forceinline__ v8f wmma_f16(v16h a, v16h b, v8f c) {
  return __builtin_amdgcn_wmma_f32_16x16x32_f16(false, a, false, b, (short)0, c,
                                                false, false);
}

// 2-stage software-pipelined K-loop, both operands pipelined (for global A+B).
__device__ __forceinline__ v8f mma_loop(const f16* A, int lda, int m0,
                                        const f16* BT, int ldb, int n0, int K) {
  v8f acc = {};
  v16h a0 = load_fragA(A, lda, m0, 0);
  v16h b0 = load_fragB(BT, ldb, n0, 0);
  for (int k0 = 32; k0 < K; k0 += 32) {
    v16h a1 = load_fragA(A, lda, m0, k0);
    v16h b1 = load_fragB(BT, ldb, n0, k0);
    acc = wmma_f16(a0, b0, acc);
    a0 = a1; b0 = b1;
  }
  return wmma_f16(a0, b0, acc);
}

// B-only pipeline: A fragments come from LDS (cheap, resident), B from global.
__device__ __forceinline__ v8f mma_loop_ldsA(const f16* A_lds, int lda,
                                             const f16* BT, int ldb, int n0, int K) {
  v8f acc = {};
  v16h b0 = load_fragB(BT, ldb, n0, 0);
  for (int k0 = 32; k0 < K; k0 += 32) {
    v16h b1 = load_fragB(BT, ldb, n0, k0);
    v16h a  = load_fragA(A_lds, lda, 0, k0 - 32);
    acc = wmma_f16(a, b0, acc);
    b0 = b1;
  }
  v16h a = load_fragA(A_lds, lda, 0, K - 32);
  return wmma_f16(a, b0, acc);
}

// ---- NT GEMM: D[M,Nc] = A[M,K](f16) * BT[Nc,K](f16)^T  --------------------
// Block = 256 threads = 8 waves as 2x4 wave-tiles -> 32x64 output tile.
template <bool BIAS, bool RELU>
__global__ void gemm_nt_wmma(const f16* __restrict__ A, const f16* __restrict__ BT,
                             const float* __restrict__ bias, float* __restrict__ D,
                             int M, int Nc, int K) {
  int wave = threadIdx.x >> 5;
  int m0 = blockIdx.x * 32 + (wave & 1) * 16;
  int n0 = blockIdx.y * 64 + (wave >> 1) * 16;
  if (m0 >= M || n0 >= Nc) return;
  v8f acc = mma_loop(A, K, m0, BT, K, n0, K);
  int lane = threadIdx.x & 31;
  int col = n0 + (lane & 15);
  int mrow = m0 + ((lane >> 4) * 8);
#pragma unroll
  for (int r = 0; r < 8; ++r) {
    float v = acc[r];
    if (BIAS) v += bias[col];
    if (RELU) v = fmaxf(v, 0.0f);
    D[(size_t)(mrow + r) * Nc + col] = v;
  }
}

// ---- kNN: per 16-row strip, WMMA Gram + register top-16 -------------------
// grid = (NN/16, NB), block = 256. Dynamic LDS:
//   aStrip[16*C] f16 | sdist[16][136] f32 | candD[16][256] f32 | candI[16][256] i32 | ssq[16]
__global__ void knn_wmma(const f16* __restrict__ Xf16, const float* __restrict__ sq,
                         int* __restrict__ knn, int C) {
  extern __shared__ char smem[];
  f16*   aStrip = (f16*)smem;
  float* sdist  = (float*)(smem + (size_t)16 * C * sizeof(f16));
  float* candD  = sdist + 16 * 136;
  int*   candI  = (int*)(candD + 16 * 256);
  float* ssq    = (float*)(candI + 16 * 256);

  int b = blockIdx.y;
  int strip0 = blockIdx.x * 16;
  const f16*   Xb  = Xf16 + (size_t)b * NN * C;
  const float* sqb = sq + (size_t)b * NN;

  for (int t = threadIdx.x; t < 16 * C; t += blockDim.x)
    aStrip[t] = Xb[(size_t)(strip0 + (t / C)) * C + (t % C)];
  if (threadIdx.x < 16) ssq[threadIdx.x] = sqb[strip0 + threadIdx.x];
  __syncthreads();

  int wave = threadIdx.x >> 5, lane = threadIdx.x & 31;
  int row = threadIdx.x >> 4, sub = threadIdx.x & 15;

  float kd[KNN_K]; int ki[KNN_K];
#pragma unroll
  for (int s = 0; s < KNN_K; ++s) { kd[s] = 1e30f; ki[s] = 0; }

  for (int cb = 0; cb < NN; cb += 128) {
    int n0 = cb + wave * 16;
    v8f acc = mma_loop_ldsA(aStrip, C, Xb, C, n0, C);
    int col = n0 + (lane & 15);
    int mr = (lane >> 4) * 8;
    float sqc = sqb[col];
#pragma unroll
    for (int r = 0; r < 8; ++r) {
      int m = mr + r;
      float d = ssq[m] + sqc - 2.0f * acc[r];
      if (col == strip0 + m) d = 1e30f;  // exclude self (loop=False)
      sdist[m * 136 + wave * 16 + (lane & 15)] = d;
    }
    __syncthreads();
    // register top-16 maintenance: 8 candidates per thread per chunk
#pragma unroll
    for (int t = 0; t < 8; ++t) {
      int cl = sub + t * 16;
      float cd = sdist[row * 136 + cl];
      int ci = cb + cl;
      if (cd < kd[KNN_K - 1]) {
#pragma unroll
        for (int s = 0; s < KNN_K; ++s) {
          if (cd < kd[s]) {
            float td = kd[s]; kd[s] = cd; cd = td;
            int   ti = ki[s]; ki[s] = ci; ci = ti;
          }
        }
      }
    }
    __syncthreads();
  }

#pragma unroll
  for (int s = 0; s < KNN_K; ++s) {
    candD[row * 256 + sub * 16 + s] = kd[s];
    candI[row * 256 + sub * 16 + s] = ki[s];
  }
  __syncthreads();
  if (threadIdx.x < 16) {
    int r = threadIdx.x;
    for (int s = 0; s < KNN_K; ++s) {
      float best = 1e31f; int bi = 0, bp = 0;
      for (int c2 = 0; c2 < 256; ++c2) {
        float d = candD[r * 256 + c2];
        if (d < best) { best = d; bi = candI[r * 256 + c2]; bp = c2; }
      }
      candD[r * 256 + bp] = 1e32f;
      knn[((size_t)b * NN + strip0 + r) * KNN_K + s] = bi;
    }
  }
}

// ---- GATv2 attention + aggregation: one wave32 per node -------------------
template <int COUT>
__global__ void gat_attention(const float* __restrict__ hl, const float* __restrict__ hr,
                              const int* __restrict__ knn, const float* __restrict__ att,
                              const float* __restrict__ bias, float* __restrict__ out) {
  constexpr int CPL = COUT / 32;
  int wave = threadIdx.x >> 5, lane = threadIdx.x & 31;
  int node = blockIdx.x * 8 + wave;                 // node in [0, NB*NN)
  int b = node / NN;
  const float* hlp = hl + (size_t)node * COUT;
  const float* hrb = hr + (size_t)b * NN * COUT;
  const int* ip = knn + (size_t)node * KNN_K;

  float hlr[CPL], ar[CPL];
#pragma unroll
  for (int c = 0; c < CPL; ++c) {
    hlr[c] = hlp[lane + 32 * c];
    ar[c]  = att[lane + 32 * c];
  }
  int idxs[KNN_K];
#pragma unroll
  for (int k = 0; k < KNN_K; ++k) idxs[k] = ip[k];

  float e[KNN_K];
  for (int k = 0; k < KNN_K; ++k) {
    const float* hp = hrb + (size_t)idxs[k] * COUT;
    float p = 0.0f;
#pragma unroll
    for (int c = 0; c < CPL; ++c) {
      float v = hlr[c] + hp[lane + 32 * c];
      v = (v > 0.0f) ? v : 0.2f * v;                // leaky_relu(0.2)
      p += ar[c] * v;
    }
#pragma unroll
    for (int off = 16; off; off >>= 1) p += __shfl_xor(p, off, 32);
    e[k] = p;
  }
  float mx = e[0];
#pragma unroll
  for (int k = 1; k < KNN_K; ++k) mx = fmaxf(mx, e[k]);
  float sum = 0.0f;
#pragma unroll
  for (int k = 0; k < KNN_K; ++k) { e[k] = __expf(e[k] - mx); sum += e[k]; }
  float inv = 1.0f / sum;

  float acc[CPL];
#pragma unroll
  for (int c = 0; c < CPL; ++c) acc[c] = 0.0f;
  for (int k = 0; k < KNN_K; ++k) {
    const float* hp = hrb + (size_t)idxs[k] * COUT;
    float al = e[k] * inv;
#pragma unroll
    for (int c = 0; c < CPL; ++c) acc[c] += al * hp[lane + 32 * c];
  }
  float* op = out + (size_t)node * COUT;
#pragma unroll
  for (int c = 0; c < CPL; ++c)
    op[lane + 32 * c] = fmaxf(acc[c] + bias[lane + 32 * c], 0.0f);
}

// ---- concat (up to 3 f32 sources) -> f16 + row sum of squares -------------
__global__ void prep_concat_f16(const float* __restrict__ s0, int w0,
                                const float* __restrict__ s1, int w1,
                                const float* __restrict__ s2, int w2,
                                f16* __restrict__ dst, float* __restrict__ sqout,
                                int M) {
  int wave = threadIdx.x >> 5, lane = threadIdx.x & 31;
  int row = blockIdx.x * 8 + wave;
  if (row >= M) return;
  int W = w0 + w1 + w2;
  float sacc = 0.0f;
  f16* dp = dst + (size_t)row * W;
  for (int c = lane; c < W; c += 32) {
    float v;
    if (c < w0)            v = s0[(size_t)row * w0 + c];
    else if (c < w0 + w1)  v = s1[(size_t)row * w1 + (c - w0)];
    else                   v = s2[(size_t)row * w2 + (c - w0 - w1)];
    dp[c] = (f16)v;
    sacc += v * v;
  }
#pragma unroll
  for (int off = 16; off; off >>= 1) sacc += __shfl_xor(sacc, off, 32);
  if (sqout != nullptr && lane == 0) sqout[row] = sacc;
}

// ---- weight transpose + f16 convert: wt[Nc][K] = (f16) w[K][Nc] -----------
__global__ void transpose_f16(const float* __restrict__ w, f16* __restrict__ wt,
                              int K, int Nc) {
  int t = blockIdx.x * blockDim.x + threadIdx.x;
  if (t >= K * Nc) return;
  int n = t / K, k = t % K;
  wt[t] = (f16)w[(size_t)k * Nc + n];
}

// ---- FeatureTransformLayer: h0[b][n][c] = sum_f mask*x*w + b --------------
__global__ void feature_transform(const float* __restrict__ x,
                                  const unsigned char* __restrict__ masks,
                                  const float* __restrict__ ftw,
                                  const float* __restrict__ ftb,
                                  float* __restrict__ h0) {
  int t = blockIdx.x * blockDim.x + threadIdx.x;
  int c = t & 255;
  int nb = t >> 8;
  int n = nb % NN, b = nb / NN;
  if (b >= NB) return;
  float s = ftb[b * 256 + c];
  const float* xr = x + (size_t)n * 16;
#pragma unroll
  for (int f = 0; f < 16; ++f) {
    if (masks[b * 16 + f]) s += xr[f] * ftw[((size_t)b * 16 + f) * 256 + c];
  }
  h0[((size_t)b * NN + n) * 256 + c] = s;
}

// ---- GeodLinear head: (h2 @ gw + gb) * tanh(SCALE*geod) -------------------
__global__ void final_geod(const float* __restrict__ h2, const float* __restrict__ gw,
                           const float* __restrict__ gb, const float* __restrict__ geod,
                           float* __restrict__ out) {
  int n = blockIdx.x * blockDim.x + threadIdx.x;
  if (n >= NN) return;
  float t = tanhf(0.5493061443340549f * geod[n]);
  const float* hp = h2 + (size_t)n * 64;
  for (int j = 0; j < 3; ++j) {
    float s = gb[j];
#pragma unroll
    for (int c = 0; c < 64; ++c) s += hp[c] * gw[c * 3 + j];
    out[n * 3 + j] = s * t;
  }
}

// ---------------------------------------------------------------------------
// Host side
// ---------------------------------------------------------------------------
namespace {

struct Scratch {
  float *h0, *o1, *o2, *o3, *o4, *hl, *hr, *h1, *h2, *sqb;
  f16 *xf16, *wtl, *wtr, *h1f16;
  int* knn;
};

inline void gemm_launch(const f16* A, const f16* BT, const float* bias, float* D,
                        int M, int Nc, int K, bool biasrelu, hipStream_t s) {
  dim3 g(M / 32, (Nc + 63) / 64);
  if (biasrelu)
    gemm_nt_wmma<true, true><<<g, 256, 0, s>>>(A, BT, bias, D, M, Nc, K);
  else
    gemm_nt_wmma<false, false><<<g, 256, 0, s>>>(A, BT, bias, D, M, Nc, K);
}

template <int COUT>
void run_layer(const Scratch& w, const float* src0, int w0, const float* src1, int w1,
               const float* wl, const float* wr, const float* att, const float* bias,
               int Cin, float* outbuf, hipStream_t s) {
  const int M = NB * NN;
  // concat -> f16 features + per-row sum of squares
  prep_concat_f16<<<M / 8, 256, 0, s>>>(src0, w0, src1, w1, nullptr, 0,
                                        w.xf16, w.sqb, M);
  // weights -> f16 transposed
  int we = Cin * COUT;
  transpose_f16<<<(we + 255) / 256, 256, 0, s>>>(wl, w.wtl, Cin, COUT);
  transpose_f16<<<(we + 255) / 256, 256, 0, s>>>(wr, w.wtr, Cin, COUT);
  // hl / hr projections (WMMA)
  gemm_launch(w.xf16, w.wtl, nullptr, w.hl, M, COUT, Cin, false, s);
  gemm_launch(w.xf16, w.wtr, nullptr, w.hr, M, COUT, Cin, false, s);
  // dynamic kNN graph via WMMA Gram
  size_t lds = (size_t)16 * Cin * sizeof(f16) + 16 * 136 * 4 + 16 * 256 * 4 +
               16 * 256 * 4 + 64;
  knn_wmma<<<dim3(NN / 16, NB), 256, lds, s>>>(w.xf16, w.sqb, w.knn, Cin);
  // attention + aggregation
  gat_attention<COUT><<<M / 8, 256, 0, s>>>(w.hl, w.hr, w.knn, att, bias, outbuf);
}

}  // namespace

extern "C" void kernel_launch(void* const* d_in, const int* in_sizes, int n_in,
                              void* d_out, int out_size, void* d_ws, size_t ws_size,
                              hipStream_t stream) {
  const float* x    = (const float*)d_in[0];
  const unsigned char* masks = (const unsigned char*)d_in[1];  // bool [3,16]
  const float* ftw  = (const float*)d_in[2];
  const float* ftb  = (const float*)d_in[3];
  const float* w1l = (const float*)d_in[4],  *w1r = (const float*)d_in[5];
  const float* a1  = (const float*)d_in[6],  *b1  = (const float*)d_in[7];
  const float* w2l = (const float*)d_in[8],  *w2r = (const float*)d_in[9];
  const float* a2  = (const float*)d_in[10], *b2  = (const float*)d_in[11];
  const float* w3l = (const float*)d_in[12], *w3r = (const float*)d_in[13];
  const float* a3  = (const float*)d_in[14], *b3  = (const float*)d_in[15];
  const float* w4l = (const float*)d_in[16], *w4r = (const float*)d_in[17];
  const float* a4  = (const float*)d_in[18], *b4  = (const float*)d_in[19];
  const float* m1w = (const float*)d_in[20], *m1b = (const float*)d_in[21];
  const float* m2w = (const float*)d_in[22], *m2b = (const float*)d_in[23];
  const float* gw  = (const float*)d_in[24], *gb  = (const float*)d_in[25];
  const float* geod = (const float*)d_in[26];
  float* out = (float*)d_out;

  const int M = NB * NN;

  // --- carve workspace (bump allocator, 256B aligned) ----------------------
  char* p = (char*)d_ws;
  auto alloc = [&](size_t bytes) -> void* {
    void* r = (void*)p;
    p += (bytes + 255) & ~(size_t)255;
    return r;
  };
  Scratch w;
  w.h0   = (float*)alloc((size_t)M * 256 * 4);
  w.o1   = (float*)alloc((size_t)M * 256 * 4);
  w.o2   = (float*)alloc((size_t)M * 512 * 4);
  w.o3   = (float*)alloc((size_t)M * 512 * 4);
  w.o4   = (float*)alloc((size_t)M * 256 * 4);
  w.hl   = (float*)alloc((size_t)M * 512 * 4);
  w.hr   = (float*)alloc((size_t)M * 512 * 4);
  w.xf16 = (f16*)alloc((size_t)M * 1024 * 2);
  w.wtl  = (f16*)alloc((size_t)1024 * 512 * 2);
  w.wtr  = (f16*)alloc((size_t)1024 * 512 * 2);
  w.sqb  = (float*)alloc((size_t)M * 4);
  w.knn  = (int*)alloc((size_t)M * KNN_K * 4);
  w.h1   = (float*)alloc((size_t)NN * 256 * 4);
  w.h1f16= (f16*)alloc((size_t)NN * 256 * 2);
  w.h2   = (float*)alloc((size_t)NN * 64 * 4);
  (void)ws_size; (void)in_sizes; (void)n_in; (void)out_size;

  // --- stage 0: feature transform -----------------------------------------
  feature_transform<<<(NB * NN * 256) / 256, 256, 0, stream>>>(x, masks, ftw, ftb, w.h0);

  // --- 4 GATv2 layers on dynamic kNN graphs -------------------------------
  run_layer<256>(w, w.h0, 256, nullptr, 0,   w1l, w1r, a1, b1, 256,  w.o1, stream);
  run_layer<512>(w, w.h0, 256, w.o1, 256,    w2l, w2r, a2, b2, 512,  w.o2, stream);
  run_layer<512>(w, w.o1, 256, w.o2, 512,    w3l, w3r, a3, b3, 768,  w.o3, stream);
  run_layer<256>(w, w.o2, 512, w.o3, 512,    w4l, w4r, a4, b4, 1024, w.o4, stream);

  // --- head: [N, 3*256] -> 256 -> 64 -> 3, geod modulation -----------------
  // gather o4 batches channel-wise: row n = [o4[0][n] | o4[1][n] | o4[2][n]]
  prep_concat_f16<<<NN / 8, 256, 0, stream>>>(
      w.o4, 256, w.o4 + (size_t)NN * 256, 256, w.o4 + (size_t)2 * NN * 256, 256,
      w.xf16, nullptr, NN);
  transpose_f16<<<(768 * 256 + 255) / 256, 256, 0, stream>>>(m1w, w.wtl, 768, 256);
  gemm_launch(w.xf16, w.wtl, m1b, w.h1, NN, 256, 768, true, stream);

  prep_concat_f16<<<NN / 8, 256, 0, stream>>>(w.h1, 256, nullptr, 0, nullptr, 0,
                                              w.h1f16, nullptr, NN);
  transpose_f16<<<(256 * 64 + 255) / 256, 256, 0, stream>>>(m2w, w.wtr, 256, 64);
  gemm_launch(w.h1f16, w.wtr, m2b, w.h2, NN, 64, 256, true, stream);

  final_geod<<<(NN + 255) / 256, 256, 0, stream>>>(w.h2, gw, gb, geod, out);
}
